// RNNTPredictorJoiner_65343632441485
// MI455X (gfx1250) — matrix operand order repaired and checked by reading, not compiled, not run on hardware
//
#include <hip/hip_runtime.h>
#include <hip/hip_bf16.h>

// RNN-T predictor+joiner fused kernel for gfx1250 (MI455X).
// logits[b,t,u,v] = sum_d tanh(enc[b,t,d] + emb[prefix[b,u],d]) * w[v,d] + bias[v]
//
// Strategy: block = 128 rows (flattened b*T*U) x full V=1024.
//   Phase 1: compute A = tanh(enc+emb) once, f16, staged in LDS in WMMA A-layout.
//   Phase 2: 8 waves x 8 vtiles each; B (w^T) streamed f32 from L2, cvt to f16;
//            v_wmma_f32_16x16x32_f16 with f32 accumulation; NT stores of output.

typedef __attribute__((ext_vector_type(16))) _Float16 v16h;
typedef __attribute__((ext_vector_type(8)))  float    v8f;

#define Bsz 8
#define Tsz 200
#define Usz 100
#define Dsz 512
#define Vsz 1024
#define MTILE 128          // rows per block
#define KCHUNKS (Dsz / 32) // 16 wmma K-steps

#if __has_builtin(__builtin_amdgcn_tanhf)
#define TANHF(x) __builtin_amdgcn_tanhf(x)
#else
#define TANHF(x) tanhf(x)
#endif

__global__ __launch_bounds__(256) void rnnt_joiner_wmma(
    const float* __restrict__ enc,     // (B,T,D)
    const int*   __restrict__ prefix,  // (B,U)
    const float* __restrict__ emb,     // (V,D)
    const float* __restrict__ w,       // (V,D)
    const float* __restrict__ bias,    // (V)
    float*       __restrict__ out)     // (B,T,U,V)
{
    // A tile in WMMA lane order: [mtile(8)][kchunk(16)][lane(32)][16 halves]
    __shared__ __align__(16) unsigned short Ash[MTILE * Dsz]; // 128 KB

    const int tid = threadIdx.x;
    const int m0  = blockIdx.x * MTILE;

    // ---------------- Phase 1: A = tanh(enc + emb[prefix]) -> f16 LDS ----------
    // 4096 slots of 16 halves; 16 slots per thread.
    #pragma unroll
    for (int it = 0; it < 16; ++it) {
        const int s     = tid + it * 256;   // 0..4095
        const int lane  = s & 31;
        const int c     = (s >> 5) & 15;    // kchunk
        const int mt    = s >> 9;           // mtile 0..7
        const int r     = lane & 15;        // row within 16x32 A tile
        const int khalf = (lane >> 4) * 8;  // K interleave select

        const int m  = m0 + mt * 16 + r;
        const int b  = m / (Tsz * Usz);
        const int tu = m - b * (Tsz * Usz);
        const int t  = tu / Usz;
        const int u  = tu - t * Usz;

        const float* encrow = enc + (size_t)(b * Tsz + t) * Dsz;
        const float* embrow = emb + (size_t)prefix[b * Usz + u] * Dsz;

        const int k0 = c * 32 + khalf;  // run of 8
        const int k1 = k0 + 16;         // run of 8

        float a[16];
        #pragma unroll
        for (int j = 0; j < 8; ++j) {
            a[j]     = TANHF(encrow[k0 + j] + embrow[k0 + j]);
            a[j + 8] = TANHF(encrow[k1 + j] + embrow[k1 + j]);
        }
        v16h h;
        #pragma unroll
        for (int j = 0; j < 16; ++j) h[j] = (_Float16)a[j];
        *(v16h*)&Ash[(size_t)(((mt * 16 + c) * 32 + lane) * 16)] = h;
    }
    __syncthreads();

    // ---------------- Phase 2: WMMA GEMM --------------------------------------
    const int lane = tid & 31;
    const int wave = tid >> 5;       // 0..7
    const int nloc = lane & 15;      // N column within 16x16 tile
    const int ksel = lane >> 4;      // B layout: 0 -> K0..7,16..23; 1 -> K8..15,24..31

    for (int vi = 0; vi < 8; ++vi) {
        const int v0 = (wave * 8 + vi) * 16;   // each wave owns 8 unique vtiles

        v8f acc[8] = {};

        for (int c = 0; c < KCHUNKS; ++c) {
            // B chunk: B[k][n] = w[v0+n][k]; lane holds two runs of 8 k's (8h.., 16+8h..).
            const float* wp = w + (size_t)(v0 + nloc) * Dsz + c * 32 + ksel * 8;
            v16h bh;
            #pragma unroll
            for (int j = 0; j < 16; ++j) bh[j] = (_Float16)wp[j < 8 ? j : j + 8];

            #pragma unroll
            for (int mt = 0; mt < 8; ++mt) {
                const v16h ah =
                    *(const v16h*)&Ash[(size_t)(((mt * 16 + c) * 32 + lane) * 16)];
                acc[mt] = __builtin_amdgcn_wmma_f32_16x16x32_f16(
                    /*neg_a=*/false, ah, /*neg_b=*/false, bh,
                    /*c_mod=*/(short)0, acc[mt],
                    /*reuse_a=*/false, /*reuse_b=*/false);
            }
        }

        // C/D layout: lane = (M>=8 ? 16:0)+N ; VGPR r = M&7.
        const float bv    = bias[v0 + nloc];
        const int   rbase = ksel * 8;
        #pragma unroll
        for (int mt = 0; mt < 8; ++mt) {
            #pragma unroll
            for (int rr = 0; rr < 8; ++rr) {
                float* p = out + (size_t)(m0 + mt * 16 + rbase + rr) * Vsz
                               + v0 + nloc;
                __builtin_nontemporal_store(acc[mt][rr] + bv, p);
            }
        }
    }
}

extern "C" void kernel_launch(void* const* d_in, const int* in_sizes, int n_in,
                              void* d_out, int out_size, void* d_ws, size_t ws_size,
                              hipStream_t stream) {
    const float* enc    = (const float*)d_in[0];  // (B,T,D)
    const int*   prefix = (const int*)  d_in[1];  // (B,U)
    const float* emb    = (const float*)d_in[2];  // (V,D)
    const float* w      = (const float*)d_in[3];  // (V,D)
    const float* bias   = (const float*)d_in[4];  // (V)
    float*       out    = (float*)d_out;          // (B,T,U,V)

    const int nrows  = Bsz * Tsz * Usz;           // 160000
    const int nblocks = nrows / MTILE;            // 1250 (exact)
    rnnt_joiner_wmma<<<dim3(nblocks), dim3(256), 0, stream>>>(
        enc, prefix, emb, w, bias, out);
}
